// DCA_78889959293548
// MI455X (gfx1250) — matrix operand order repaired and checked by reading, not compiled, not run on hardware
//
#include <hip/hip_runtime.h>

// ---------------------------------------------------------------- types
typedef __attribute__((ext_vector_type(16))) __bf16 v16bf;
typedef __attribute__((ext_vector_type(8)))  float  v8f;

// TDM descriptor group types (6-arg builtin: g0 u32x4, g1 i32x8, g2 i32x4, g3 i32x4, i32x8, cpol)
typedef __attribute__((ext_vector_type(4))) unsigned int tdm_g0_t;
typedef __attribute__((ext_vector_type(8))) int          tdm_g1_t;
typedef __attribute__((ext_vector_type(4))) int          tdm_g2_t;

static constexpr int kB = 128;
static constexpr int kN = 192;
static constexpr int kS = 1920;
static constexpr int FEAT[4] = {128, 256, 512, 1024};
static constexpr int OFF[4]  = {0, 128, 384, 896};

// ---------------------------------------------------------------- wmma helpers
__device__ __forceinline__ v8f wmma_bf16(v16bf a, v16bf b, v8f c) {
  // D = A(16x32) * B(32x16) + C, fp32 accumulate
  return __builtin_amdgcn_wmma_f32_16x16x32_bf16(false, a, false, b, (short)0, c,
                                                 false, false);
}

// A-matrix fragment (M=16, K=32) from row-major [M][ld] bf16 LDS tile.
__device__ __forceinline__ v16bf fragA(const __bf16* base, int ld, int kBase, int lane) {
  const int m  = lane & 15;
  const int hk = (lane >> 4) & 1;
  const __bf16* p = base + m * ld + kBase + hk * 8;
  v16bf f;
#pragma unroll
  for (int j = 0; j < 8; ++j) f[j] = p[j];
#pragma unroll
  for (int j = 0; j < 8; ++j) f[8 + j] = p[16 + j];
  return f;
}

// B-matrix fragment (K=32, N=16) from B^T storage: row-major [N][ld] bf16 LDS.
__device__ __forceinline__ v16bf fragB(const __bf16* base, int ld, int kBase, int lane) {
  const int n  = lane & 15;
  const int hk = (lane >> 4) & 1;
  const __bf16* p = base + n * ld + kBase + hk * 8;
  v16bf f;
#pragma unroll
  for (int j = 0; j < 8; ++j) f[j] = p[j];
#pragma unroll
  for (int j = 0; j < 8; ++j) f[8 + j] = p[16 + j];
  return f;
}

__device__ __forceinline__ float half16_max(float v) {
  v = fmaxf(v, __shfl_xor(v, 1, 32));
  v = fmaxf(v, __shfl_xor(v, 2, 32));
  v = fmaxf(v, __shfl_xor(v, 4, 32));
  v = fmaxf(v, __shfl_xor(v, 8, 32));
  return v;
}
__device__ __forceinline__ float half16_sum(float v) {
  v += __shfl_xor(v, 1, 32);
  v += __shfl_xor(v, 2, 32);
  v += __shfl_xor(v, 4, 32);
  v += __shfl_xor(v, 8, 32);
  return v;
}

// ---------------------------------------------------------------- TDM 2-D tile load
// Async DMA of a (tileY x tileX) bf16 tile (row stride = stride0 elements) from
// global memory into contiguous LDS at ldsOff. Tracked by TENSORcnt.
__device__ __forceinline__ void tdm_load_2d_bf16(const __bf16* gsrc, unsigned ldsOff,
                                                 unsigned tileX, unsigned tileY,
                                                 unsigned tensor0, unsigned tensor1,
                                                 unsigned long long stride0) {
  const unsigned long long ga = (unsigned long long)(uintptr_t)gsrc;
  tdm_g0_t g0;
  g0[0] = 1u;                                                   // count=1, user mode
  g0[1] = ldsOff;                                               // lds_addr [63:32]
  g0[2] = (unsigned)(ga & 0xFFFFFFFFull);                       // global_addr lo
  g0[3] = (unsigned)((ga >> 32) & 0x1FFFFFFull) | (2u << 30);   // addr hi | type=2
  tdm_g1_t g1;
  g1[0] = (int)(1u << 16);                                      // data_size = 2 bytes
  g1[1] = (int)((tensor0 & 0xFFFFu) << 16);                     // tensor_dim0 lo16
  g1[2] = (int)((tensor0 >> 16) | ((tensor1 & 0xFFFFu) << 16)); // dim0 hi | dim1 lo
  g1[3] = (int)((tensor1 >> 16) | (tileX << 16));               // dim1 hi | tile_dim0
  g1[4] = (int)tileY;                                           // tile_dim1 (dim2=0)
  g1[5] = (int)(unsigned)(stride0 & 0xFFFFFFFFull);             // dim0 stride lo32
  g1[6] = (int)(unsigned)((stride0 >> 32) & 0xFFFFull);         // stride hi16
  g1[7] = 0;
  tdm_g2_t z;
  z[0] = 0; z[1] = 0; z[2] = 0; z[3] = 0;
  tdm_g1_t z8;
#pragma unroll
  for (int j = 0; j < 8; ++j) z8[j] = 0;
  __builtin_amdgcn_tensor_load_to_lds(g0, g1, z, z, z8, 0);
}

// ---------------------------------------------------------------- LayerNorm
struct LNArgs {
  const float* x[4];
  const float* xres_in;
  const float* g[4];
  const float* b[4];
  float*  xres_out;
  __bf16* xcb;
};

template <int STAGE>
__global__ void __launch_bounds__(256) ln_kernel(LNArgs a) {
  const int bn = blockIdx.x;
  __shared__ float red0[256];
  __shared__ float red1[256];
#pragma unroll
  for (int i = 0; i < 4; ++i) {
    const int C = FEAT[i];
    const float* row = (STAGE == 0) ? (a.x[i] + (size_t)bn * C)
                                    : (a.xres_in + (size_t)bn * kS + OFF[i]);
    float s = 0.f, sq = 0.f;
    for (int c = threadIdx.x; c < C; c += 256) {
      const float v = row[c];
      s += v; sq += v * v;
    }
    red0[threadIdx.x] = s; red1[threadIdx.x] = sq;
    __syncthreads();
    for (int st = 128; st > 0; st >>= 1) {
      if (threadIdx.x < st) {
        red0[threadIdx.x] += red0[threadIdx.x + st];
        red1[threadIdx.x] += red1[threadIdx.x + st];
      }
      __syncthreads();
    }
    const float mean = red0[0] / (float)C;
    const float var  = red1[0] / (float)C - mean * mean;
    const float inv  = rsqrtf(var + 1e-6f);
    for (int c = threadIdx.x; c < C; c += 256) {
      const float v = row[c];
      if (STAGE == 0) a.xres_out[(size_t)bn * kS + OFF[i] + c] = v;
      a.xcb[(size_t)bn * kS + OFF[i] + c] =
          (__bf16)((v - mean) * inv * a.g[i][c] + a.b[i][c]);
    }
    __syncthreads();
  }
}

// ---------------------------------------------------------------- channel attention
// Flash-style over e (S=1920) in 32-wide tiles, double-buffered TDM prefetch.
struct CAArgs {
  const __bf16* xcb;
  float* xres;
  const float *wq, *bq, *wk, *bk, *wv, *bv, *wp, *bp;
  int Cq, coff, nDT;
};

__global__ void __launch_bounds__(64) chan_att(CAArgs a) {
  const int dt = blockIdx.x % a.nDT;
  const int b  = blockIdx.x / a.nDT;
  const int tid = threadIdx.x, lane = tid & 31, wave = tid >> 5;
  constexpr int nT = kS / 32;

  __shared__ __bf16 Xs[2][192][32]; // raw TDM staging tiles [n][e]
  __shared__ __bf16 Qs[16][200];    // [d][n], dw(q) applied
  __shared__ __bf16 Ks[32][200];    // [e][n], dw(k) applied  (B^T for logits)
  __shared__ __bf16 Vs[192][40];    // [n][e], dw(v) applied  (B^T for P*V)
  __shared__ __bf16 Ps[16][40];     // [d][e] probabilities
  __shared__ float  maxw[2][16], sums[2][16], mrow[16], srow[16], corr[16];

  const __bf16* tileBase = a.xcb + ((size_t)b * kN) * kS;

  // prologue: kick off DMA of tile 0 while Q is staged
  if (wave == 0) {
    tdm_load_2d_bf16(tileBase, (unsigned)(uintptr_t)&Xs[0][0][0],
                     32u, 192u, (unsigned)kS, (unsigned)kN,
                     (unsigned long long)kS);
  }

  for (int idx = tid; idx < 16 * 192; idx += 64) {
    const int d = idx / 192, n = idx % 192;
    const int c = dt * 16 + d;
    const float x = (float)a.xcb[((size_t)b * kN + n) * kS + a.coff + c];
    Qs[d][n] = (__bf16)(x * a.wq[c] + a.bq[c]);
  }
  if (tid < 16) { mrow[tid] = -3.0e38f; srow[tid] = 0.f; }

  v8f o[6];
#pragma unroll
  for (int t = 0; t < 6; ++t)
#pragma unroll
    for (int j = 0; j < 8; ++j) o[t][j] = 0.f;

  const float rs = rsqrtf((float)kS);

  for (int et = 0; et < nT; ++et) {
    const int eBase = et * 32;
    const int buf = et & 1;
    // issue DMA for next tile, then wait for current tile to land
    if (wave == 0) {
      if (et + 1 < nT) {
        tdm_load_2d_bf16(tileBase + (size_t)(et + 1) * 32,
                         (unsigned)(uintptr_t)&Xs[(et + 1) & 1][0][0],
                         32u, 192u, (unsigned)kS, (unsigned)kN,
                         (unsigned long long)kS);
        __builtin_amdgcn_s_wait_tensorcnt(1);
      } else {
        __builtin_amdgcn_s_wait_tensorcnt(0);
      }
    }
    __syncthreads();

    // repack staged tile -> compute layouts with depthwise scale/bias fused
    for (int idx = tid; idx < 32 * 192; idx += 64) {
      const int n = idx >> 5, e = idx & 31;
      const int eg = eBase + e;
      const float x = (float)Xs[buf][n][e];
      Ks[e][n] = (__bf16)(x * a.wk[eg] + a.bk[eg]);
      Vs[n][e] = (__bf16)(x * a.wv[eg] + a.bv[eg]);
    }
    __syncthreads();

    // logits 16x16 per wave (e sub-tile = wave*16), K-dim = tokens (192)
    v8f l;
#pragma unroll
    for (int j = 0; j < 8; ++j) l[j] = 0.f;
    for (int kk = 0; kk < 192; kk += 32) {
      v16bf af  = fragA(&Qs[0][0], 200, kk, lane);
      v16bf bfr = fragB(&Ks[wave * 16][0], 200, kk, lane);
      l = wmma_bf16(af, bfr, l);
    }
#pragma unroll
    for (int j = 0; j < 8; ++j) l[j] *= rs;
#pragma unroll
    for (int j = 0; j < 8; ++j) {
      const float rm = half16_max(l[j]);
      if ((lane & 15) == 0) maxw[wave][j + 8 * (lane >> 4)] = rm;
    }
    __syncthreads();
    if (tid < 16) {
      const float mo = mrow[tid];
      const float mn = fmaxf(mo, fmaxf(maxw[0][tid], maxw[1][tid]));
      const float cf = __expf(mo - mn);
      corr[tid] = cf;
      srow[tid] *= cf;
      mrow[tid] = mn;
    }
    __syncthreads();
#pragma unroll
    for (int j = 0; j < 8; ++j) {
      const int m = j + 8 * (lane >> 4);
      const float p = __expf(l[j] - mrow[m]);
      Ps[m][wave * 16 + (lane & 15)] = (__bf16)p;
      const float ps = half16_sum(p);
      if ((lane & 15) == 0) sums[wave][m] = ps;
    }
    __syncthreads();
    if (tid < 16) srow[tid] += sums[0][tid] + sums[1][tid];

    // O[d, n] += P[d, e(32)] * V[e, n];  2 waves x 6 n-tiles cover N=192
#pragma unroll
    for (int t = 0; t < 6; ++t) {
      const int nt = wave * 6 + t;
#pragma unroll
      for (int j = 0; j < 8; ++j) {
        const int m = j + 8 * (lane >> 4);
        o[t][j] *= corr[m];
      }
      v16bf pa = fragA(&Ps[0][0], 40, 0, lane);
      v16bf vb = fragB(&Vs[nt * 16][0], 40, 0, lane);
      o[t] = wmma_bf16(pa, vb, o[t]);
    }
  }
  __syncthreads();
#pragma unroll
  for (int t = 0; t < 6; ++t) {
    const int nt = wave * 6 + t;
#pragma unroll
    for (int j = 0; j < 8; ++j) {
      const int m = j + 8 * (lane >> 4);
      const int d = dt * 16 + m;
      const int n = nt * 16 + (lane & 15);
      const float val = (o[t][j] / srow[m]) * a.wp[d] + a.bp[d];
      a.xres[((size_t)b * kN + n) * kS + a.coff + d] += val;
    }
  }
}

// ---------------------------------------------------------------- spatial attention
struct SAArgs {
  const __bf16* xcb;
  const float* xres;
  float* outp;
  const float *wq, *bq, *wk, *bk, *wv, *bv, *wp, *bp;
  int Ci, coff, Ev;
};

__global__ void __launch_bounds__(128) spat_att(SAArgs a) {
  int blk = blockIdx.x;
  const int nt = blk % 12; blk /= 12;
  const int h  = blk % 4;
  const int b  = blk / 4;
  const int tid = threadIdx.x, lane = tid & 31, wave = tid >> 5;

  __shared__ __bf16 Qs[16][40];
  __shared__ __bf16 Km[192][40];
  __shared__ __bf16 Ps[16][200];
  __shared__ __bf16 Vs[64][200];
  __shared__ float  maxw[4][16], sums[4][16], mrow[16], srow[16];

  const int dOff = h * (kS / 4);
  v8f acc[3];
#pragma unroll
  for (int t = 0; t < 3; ++t)
#pragma unroll
    for (int j = 0; j < 8; ++j) acc[t][j] = 0.f;

  for (int kb = 0; kb < kS / 4; kb += 32) {
    __syncthreads();
    for (int idx = tid; idx < 16 * 32; idx += 128) {
      const int r = idx >> 5, c = idx & 31;
      const int n = nt * 16 + r, ch = dOff + kb + c;
      const float x = (float)a.xcb[((size_t)b * kN + n) * kS + ch];
      Qs[r][c] = (__bf16)(x * a.wq[ch] + a.bq[ch]);
    }
    for (int idx = tid; idx < 192 * 32; idx += 128) {
      const int r = idx >> 5, c = idx & 31;
      const int ch = dOff + kb + c;
      const float x = (float)a.xcb[((size_t)b * kN + r) * kS + ch];
      Km[r][c] = (__bf16)(x * a.wk[ch] + a.bk[ch]);
    }
    if (kb + 32 < kS / 4) {
      __builtin_prefetch(&a.xcb[((size_t)b * kN + (tid % 192)) * kS + dOff + kb + 32], 0, 1);
    }
    __syncthreads();
#pragma unroll
    for (int t = 0; t < 3; ++t) {
      const int mBase = wave * 48 + t * 16;
      v16bf af  = fragA(&Qs[0][0], 40, 0, lane);
      v16bf bfr = fragB(&Km[mBase][0], 40, 0, lane);
      acc[t] = wmma_bf16(af, bfr, acc[t]);
    }
  }

  const float rs = rsqrtf((float)(kS / 4));
#pragma unroll
  for (int j = 0; j < 8; ++j) {
    float rm = -3.0e38f;
#pragma unroll
    for (int t = 0; t < 3; ++t) { acc[t][j] *= rs; rm = fmaxf(rm, acc[t][j]); }
    rm = half16_max(rm);
    if ((lane & 15) == 0) maxw[wave][j + 8 * (lane >> 4)] = rm;
  }
  __syncthreads();
  if (tid < 16)
    mrow[tid] = fmaxf(fmaxf(maxw[0][tid], maxw[1][tid]),
                      fmaxf(maxw[2][tid], maxw[3][tid]));
  __syncthreads();

  float lsum[8];
#pragma unroll
  for (int j = 0; j < 8; ++j) lsum[j] = 0.f;
#pragma unroll
  for (int t = 0; t < 3; ++t) {
#pragma unroll
    for (int j = 0; j < 8; ++j) {
      const int r = j + 8 * (lane >> 4);
      const float p = __expf(acc[t][j] - mrow[r]);
      Ps[r][wave * 48 + t * 16 + (lane & 15)] = (__bf16)p;
      lsum[j] += p;
    }
  }
#pragma unroll
  for (int j = 0; j < 8; ++j) {
    const float s = half16_sum(lsum[j]);
    if ((lane & 15) == 0) sums[wave][j + 8 * (lane >> 4)] = s;
  }
  __syncthreads();
  if (tid < 16)
    srow[tid] = sums[0][tid] + sums[1][tid] + sums[2][tid] + sums[3][tid];

  const int nChunks = (a.Ev + 63) / 64;
  for (int ec = 0; ec < nChunks; ++ec) {
    const int eBase = ec * 64;
    __syncthreads();
    for (int idx = tid; idx < 64 * 192; idx += 128) {
      const int m = idx >> 6, e = idx & 63;
      const int eg = eBase + e;
      float x = 0.f;
      if (eg < a.Ev) {
        const int cs = h * a.Ev + eg;
        x = (float)a.xcb[((size_t)b * kN + m) * kS + a.coff + cs];
        x = x * a.wv[cs] + a.bv[cs];
      }
      Vs[e][m] = (__bf16)x;
    }
    __syncthreads();
    if (eBase + wave * 16 < a.Ev) {
      v8f o;
#pragma unroll
      for (int j = 0; j < 8; ++j) o[j] = 0.f;
      for (int mm = 0; mm < 192; mm += 32) {
        v16bf pa = fragA(&Ps[0][0], 200, mm, lane);
        v16bf vb = fragB(&Vs[wave * 16][0], 200, mm, lane);
        o = wmma_bf16(pa, vb, o);
      }
#pragma unroll
      for (int j = 0; j < 8; ++j) {
        const int r = j + 8 * (lane >> 4);
        const int n = nt * 16 + r;
        const int cs = h * a.Ev + eBase + wave * 16 + (lane & 15);
        float val = (o[j] / srow[r]) * a.wp[cs] + a.bp[cs];
        val += a.xres[((size_t)b * kN + n) * kS + a.coff + cs];
        a.outp[((size_t)b * kN + n) * a.Ci + cs] = val;
      }
    }
  }
}

// ---------------------------------------------------------------- host launch
extern "C" void kernel_launch(void* const* d_in, const int* in_sizes, int n_in,
                              void* d_out, int out_size, void* d_ws, size_t ws_size,
                              hipStream_t stream) {
  auto f = [&](int i) { return (const float*)d_in[i]; };
  const int CLN = 4, SLN = 12, CATT = 20, SATT = 52;

  const size_t nTok = (size_t)kB * kN;
  const size_t nEl  = nTok * kS;
  float*  xres = (float*)d_ws;
  __bf16* xcb  = (__bf16*)((char*)d_ws + nEl * sizeof(float));

  {
    LNArgs a{};
    for (int i = 0; i < 4; ++i) {
      a.x[i] = f(i);
      a.g[i] = f(CLN + 2 * i);
      a.b[i] = f(CLN + 2 * i + 1);
    }
    a.xres_in = nullptr;
    a.xres_out = xres;
    a.xcb = xcb;
    ln_kernel<0><<<kB * kN, 256, 0, stream>>>(a);
  }

  for (int i = 0; i < 4; ++i) {
    const int base = CATT + 8 * i;
    CAArgs a{xcb, xres,
             f(base + 0), f(base + 1), f(base + 2), f(base + 3),
             f(base + 4), f(base + 5), f(base + 6), f(base + 7),
             FEAT[i], OFF[i], FEAT[i] / 16};
    chan_att<<<kB * (FEAT[i] / 16), 64, 0, stream>>>(a);
  }

  {
    LNArgs a{};
    for (int i = 0; i < 4; ++i) {
      a.x[i] = nullptr;
      a.g[i] = f(SLN + 2 * i);
      a.b[i] = f(SLN + 2 * i + 1);
    }
    a.xres_in = xres;
    a.xres_out = nullptr;
    a.xcb = xcb;
    ln_kernel<1><<<kB * kN, 256, 0, stream>>>(a);
  }

  for (int i = 0; i < 4; ++i) {
    const int base = SATT + 8 * i;
    float* outi = (float*)d_out + nTok * OFF[i];
    SAArgs a{xcb, xres, outi,
             f(base + 0), f(base + 1), f(base + 2), f(base + 3),
             f(base + 4), f(base + 5), f(base + 6), f(base + 7),
             FEAT[i], OFF[i], FEAT[i] / 4};
    spat_att<<<kB * 4 * 12, 128, 0, stream>>>(a);
  }
}
